// NeuralCDE_45621142618658
// MI455X (gfx1250) — matrix operand order repaired and checked by reading, not compiled, not run on hardware
//
#include <hip/hip_runtime.h>
#include <math.h>

// ---------------------------------------------------------------------------
// Neural CDE persistent kernel for MI455X (gfx1250, wave32, WMMA f16).
// Grid: 4 workgroups x 256 threads (8 wave32). Each workgroup owns 16 batch
// rows and runs the entire 255-step / 6-stage RK recurrence with all vector-
// field weights resident in LDS as f16 (311 KB of the 320 KB WGP LDS).
// Transcendentals use raw gfx1250 hardware ops (v_exp/v_log/v_tanh),
// branchless, to keep the serial latency chain short.
// ---------------------------------------------------------------------------

typedef __attribute__((ext_vector_type(16))) _Float16 v16h;
typedef __attribute__((ext_vector_type(2)))  _Float16 h2v;
typedef __attribute__((ext_vector_type(8)))  float    v8f;

#define WMMA_F16(a, b, c) \
  __builtin_amdgcn_wmma_f32_16x16x32_f16(false, (a), false, (b), (short)0, (c), false, false)

// ---- problem sizes ---------------------------------------------------------
#define NB   64
#define NT   256
#define NC   16
#define NS   64
#define NH   128

// ---- LDS layout (bytes) ----------------------------------------------------
#define OFF_W0  0          // vf_w0 f16 [128][64]   16384
#define OFF_W1  16384      // vf_w1 f16 [128][128]  32768
#define OFF_W2  49152      // vf_w2 f16 [1024][128] 262144  (ic weights live here first)
#define OFF_B0  311296     // b0 f16 [128]          256
#define OFF_B1  311552     // b1 f16 [128]          256
#define OFF_B2  311808     // b2 f16 [1024]         2048
#define OFF_YT  313856     // stage input f16 [16][64]      2048
#define OFF_A1  315904     // h1 f16 [16][128] / f32 staging [16][64]  4096
#define OFF_A2  320000     // h2 f16 [16][128]      4096
#define OFF_DX  324096     // dX f32 [16][16]       1024
#define OFF_RW  325120     // ro_w f32 [6][64]      1536
#define OFF_RB  326656     // ro_b f32 [6] (pad)    32
#define OFF_O6  326688     // out6 f32 [16][6]      384
#define SMEM_BYTES 327072

// ---- RK tableau ------------------------------------------------------------
__constant__ float ACO[6][5] = {
  {0.f, 0.f, 0.f, 0.f, 0.f},
  {0.161f, 0.f, 0.f, 0.f, 0.f},
  {-0.008480655492356989f, 0.335480655492357f, 0.f, 0.f, 0.f},
  {2.8971530571054935f, -6.359448489975075f, 4.3622954328695815f, 0.f, 0.f},
  {5.325864828439257f, -11.748883564062828f, 7.4955393428898365f, -0.09249506636175525f, 0.f},
  {5.86145544294642f, -12.92096931784711f, 8.159367898576159f, -0.071584973281401f, -0.028269050394068383f},
};
__constant__ float SCC[6] = {0.f, 0.161f, 0.327f, 0.9f, 0.9800255409045097f, 1.f};
__constant__ float BWc[6] = {0.09646076681806523f, 0.01f, 0.4798896504144996f,
                             1.379008574103742f, -3.290069515436081f, 2.324710524099774f};

// ---- fast hw transcendentals (branchless) ----------------------------------
__device__ __forceinline__ float fast_exp2(float x) {
#if defined(__has_builtin) && __has_builtin(__builtin_amdgcn_exp2f)
  return __builtin_amdgcn_exp2f(x);
#else
  return exp2f(x);
#endif
}
__device__ __forceinline__ float fast_log2(float x) {
#if defined(__has_builtin) && __has_builtin(__builtin_amdgcn_logf)
  return __builtin_amdgcn_logf(x);
#else
  return log2f(x);
#endif
}
__device__ __forceinline__ float fast_rcp(float x) {
#if defined(__has_builtin) && __has_builtin(__builtin_amdgcn_rcpf)
  return __builtin_amdgcn_rcpf(x);
#else
  return 1.0f / x;
#endif
}
__device__ __forceinline__ float fast_tanh(float x) {
#if defined(__has_builtin) && __has_builtin(__builtin_amdgcn_tanhf)
  return __builtin_amdgcn_tanhf(x);   // gfx1250 V_TANH_F32
#else
  const float ax = __builtin_fabsf(x);
  const float e  = fast_exp2(-2.8853901817f * ax);   // exp(-2|x|)
  const float r  = (1.f - e) * fast_rcp(1.f + e);
  return __builtin_copysignf(r, x);
#endif
}
// softplus(x) = max(x,0) + ln2 * log2(1 + exp2(-log2e*|x|)), branchless
__device__ __forceinline__ float softplus_f(float x) {
  const float ax = __builtin_fabsf(x);
  const float e  = fast_exp2(-1.44269504089f * ax);
  const float l  = 0.69314718056f * fast_log2(1.f + e);
  return __builtin_fmaxf(x, 0.f) + l;
}

// A fragment (16x32 f16) from LDS row-major [M=16][ldk], starting column k0.
// wave32 layout: lane(0-15)->M, halves hold K = 16*half .. 16*half+15 (contiguous).
__device__ __forceinline__ v16h load_A32(const _Float16* a, int ldk, int k0, int lane) {
  const int m  = lane & 15;
  const int hh = lane >> 4;
  v16h r;
#pragma unroll
  for (int p = 0; p < 8; ++p) {
    const int kb = 2 * p + 16 * hh + k0;
    h2v t = *(const h2v*)(a + m * ldk + kb);
    r[2 * p]     = t.x;
    r[2 * p + 1] = t.y;
  }
  return r;
}

// B fragment (32x16 f16) from LDS weight matrix row-major [N][ldk] (= W[n][k]).
// lane(0-15)->N column, halves hold K = 16*half .. 16*half+15 (contiguous).
__device__ __forceinline__ v16h load_B32(const _Float16* w, int ldk, int n0, int k0, int lane) {
  const int n  = n0 + (lane & 15);
  const int hh = lane >> 4;
  return *(const v16h*)(w + n * ldk + k0 + 16 * hh);
}

__global__ __launch_bounds__(256, 1) void neural_cde_kernel(
    const float* __restrict__ xs,
    const float* __restrict__ icw0, const float* __restrict__ icb0,
    const float* __restrict__ icw1, const float* __restrict__ icb1,
    const float* __restrict__ icw2, const float* __restrict__ icb2,
    const float* __restrict__ vfw0, const float* __restrict__ vfb0,
    const float* __restrict__ vfw1, const float* __restrict__ vfb1,
    const float* __restrict__ vfw2, const float* __restrict__ vfb2,
    const float* __restrict__ rowt, const float* __restrict__ robv,
    float* __restrict__ out)
{
  extern __shared__ char smem[];
  _Float16* w0h  = (_Float16*)(smem + OFF_W0);
  _Float16* w1h  = (_Float16*)(smem + OFF_W1);
  _Float16* w2h  = (_Float16*)(smem + OFF_W2);
  _Float16* b0h  = (_Float16*)(smem + OFF_B0);
  _Float16* b1h  = (_Float16*)(smem + OFF_B1);
  _Float16* b2h  = (_Float16*)(smem + OFF_B2);
  _Float16* ytmp = (_Float16*)(smem + OFF_YT);   // [16][64] stage input (f16)
  _Float16* a1   = (_Float16*)(smem + OFF_A1);   // [16][128] h1
  _Float16* a2   = (_Float16*)(smem + OFF_A2);   // [16][128] h2
  float*    fstg = (float*)   (smem + OFF_A1);   // [16][64] f32 staging (time-shares h1)
  float*    dxs  = (float*)   (smem + OFF_DX);   // [16][16]
  float*    roW  = (float*)   (smem + OFF_RW);
  float*    roB  = (float*)   (smem + OFF_RB);
  float*    o6   = (float*)   (smem + OFF_O6);

  // ic weights transiently occupy the (large) w2 region
  _Float16* icw0p = w2h;                    // [128][32] (K padded 16->32 with zeros)
  _Float16* icw1p = w2h + 128 * 32;         // [128][128]
  _Float16* icw2p = icw1p + 128 * 128;      // [64][128]

  const int tid   = threadIdx.x;
  const int lane  = tid & 31;
  const int wv    = tid >> 5;      // wave 0..7
  const int hh    = lane >> 4;     // half-wave
  const int bcol  = lane & 15;
  const int bbase = blockIdx.x * 16;
  const float hstep = 1.0f / (float)(NT - 1);
  const float invh  = (float)(NT - 1);

  // per-step readout: y regs -> fstg -> 6-dim linear -> Gram-Schmidt -> out
  float y[4];
  auto readout = [&](int tout) {
    __syncthreads();
#pragma unroll
    for (int q = 0; q < 4; ++q)
      fstg[bcol * NS + wv * 8 + hh * 4 + q] = y[q];
    __syncthreads();
    if (tid < 96) {
      const int b = tid / 6, j = tid % 6;
      float o = roB[j];
#pragma unroll 8
      for (int s = 0; s < NS; ++s) o += fstg[b * NS + s] * roW[j * NS + s];
      o6[b * 6 + j] = o;
    }
    __syncthreads();
    if (tid < 16) {
      const float* v = o6 + tid * 6;
      const float a1x = v[0], a1y = v[1], a1z = v[2];
      const float a2x = v[3], a2y = v[4], a2z = v[5];
      const float n1 = rsqrtf(a1x * a1x + a1y * a1y + a1z * a1z);
      const float b1x = a1x * n1, b1y = a1y * n1, b1z = a1z * n1;
      const float d  = b1x * a2x + b1y * a2y + b1z * a2z;
      const float px = a2x - d * b1x, py = a2y - d * b1y, pz = a2z - d * b1z;
      const float n2 = rsqrtf(px * px + py * py + pz * pz);
      const float b2x = px * n2, b2y = py * n2, b2z = pz * n2;
      const float b3x = b1y * b2z - b1z * b2y;
      const float b3y = b1z * b2x - b1x * b2z;
      const float b3z = b1x * b2y - b1y * b2x;
      float* o = out + ((size_t)(bbase + tid) * NT + tout) * 9;
      o[0] = b1x; o[1] = b2x; o[2] = b3x;
      o[3] = b1y; o[4] = b2y; o[5] = b3y;
      o[6] = b1z; o[7] = b2z; o[8] = b3z;
    }
  };

  // ------------------------- IC phase (y0 = ic-MLP(x[:,0])) -----------------
  for (int i = tid; i < 128 * 32; i += 256) {
    const int n = i >> 5, k = i & 31;
    icw0p[i] = (_Float16)((k < NC) ? icw0[n * NC + k] : 0.f);
  }
  for (int i = tid; i < 128 * 128; i += 256) icw1p[i] = (_Float16)icw1[i];
  for (int i = tid; i < 64 * 128;  i += 256) icw2p[i] = (_Float16)icw2[i];
  for (int i = tid; i < 128; i += 256) { b0h[i] = (_Float16)icb0[i]; b1h[i] = (_Float16)icb1[i]; }
  for (int i = tid; i < 64;  i += 256) b2h[i] = (_Float16)icb2[i];
  // x0 -> ytmp [16][32] (K padded)
  for (int i = tid; i < 16 * 32; i += 256) {
    const int m = i >> 5, k = i & 31;
    ytmp[i] = (_Float16)((k < NC) ? xs[((size_t)(bbase + m) * NT + 0) * NC + k] : 0.f);
  }
  for (int i = tid; i < 6 * NS; i += 256) roW[i] = rowt[i];
  if (tid < 6) roB[tid] = robv[tid];
  __syncthreads();

  { // ic L1: (16x32)x(32x16) per wave, N tile = wv*16
    v8f acc = {};
    acc = WMMA_F16(load_A32(ytmp, 32, 0, lane), load_B32(icw0p, 32, wv * 16, 0, lane), acc);
    const int n = wv * 16 + bcol;
#pragma unroll
    for (int r = 0; r < 8; ++r)
      a1[(r + 8 * hh) * NH + n] = (_Float16)softplus_f(acc[r] + (float)b0h[n]);
  }
  __syncthreads();
  { // ic L2
    v8f acc = {};
#pragma unroll
    for (int ks = 0; ks < 4; ++ks)
      acc = WMMA_F16(load_A32(a1, NH, ks * 32, lane), load_B32(icw1p, NH, wv * 16, ks * 32, lane), acc);
    const int n = wv * 16 + bcol;
#pragma unroll
    for (int r = 0; r < 8; ++r)
      a2[(r + 8 * hh) * NH + n] = (_Float16)softplus_f(acc[r] + (float)b1h[n]);
  }
  __syncthreads();
  if (wv < 4) { // ic L3 (N=64): identity activation -> f32 staging
    v8f acc = {};
#pragma unroll
    for (int ks = 0; ks < 4; ++ks)
      acc = WMMA_F16(load_A32(a2, NH, ks * 32, lane), load_B32(icw2p, NH, wv * 16, ks * 32, lane), acc);
    const int n = wv * 16 + bcol;
#pragma unroll
    for (int r = 0; r < 8; ++r)
      fstg[(r + 8 * hh) * NS + n] = acc[r] + (float)b2h[n];
  }
  __syncthreads();
#pragma unroll
  for (int q = 0; q < 4; ++q) y[q] = fstg[bcol * NS + wv * 8 + hh * 4 + q];
  readout(0);
  __syncthreads();

  // ------------------------- load vector-field weights ----------------------
  for (int i = tid; i < NH * NS;       i += 256) w0h[i] = (_Float16)vfw0[i];
  for (int i = tid; i < NH * NH;       i += 256) w1h[i] = (_Float16)vfw1[i];
  for (int i = tid; i < NS * NC * NH;  i += 256) w2h[i] = (_Float16)vfw2[i];
  for (int i = tid; i < NH;  i += 256) { b0h[i] = (_Float16)vfb0[i]; b1h[i] = (_Float16)vfb1[i]; }
  for (int i = tid; i < NS * NC; i += 256) b2h[i] = (_Float16)vfb2[i];
  __syncthreads();

  // ------------------------- RK45 time scan ---------------------------------
  float kk[6][4];
  for (int t = 0; t < NT - 1; ++t) {
#pragma unroll
    for (int st = 0; st < 6; ++st) {
      // stage input y_stage -> ytmp (f16), each lane owns 4 (b,s) entries
#pragma unroll
      for (int q = 0; q < 4; ++q) {
        float ysv = y[q];
#pragma unroll
        for (int j = 0; j < st; ++j) ysv += hstep * ACO[st][j] * kk[j][q];
        ytmp[bcol * NS + wv * 8 + hh * 4 + q] = (_Float16)ysv;
      }
      { // dXdt(sc) for this stage: one value per (b,c) thread
        const int b = tid >> 4, c = tid & 15;
        const size_t g = ((size_t)(bbase + b) * NT + t) * NC + c;
        const float xi   = xs[g];
        const float xip1 = xs[g + NC];
        const float dip1 = (xip1 - xi) * invh;
        const float di   = (t == 0) ? dip1 : (xi - xs[g - NC]) * invh;
        const float sc = SCC[st], s2 = sc * sc;
        dxs[tid] = (6.f * s2 - 6.f * sc) * invh * xi + (3.f * s2 - 4.f * sc + 1.f) * di
                 + (6.f * sc - 6.f * s2) * invh * xip1 + (3.f * s2 - 2.f * sc) * dip1;
      }
      __syncthreads();
      { // L1: y(16x64) x W0^T -> h1 (softplus)
        v8f acc = {};
#pragma unroll
        for (int ks = 0; ks < 2; ++ks)
          acc = WMMA_F16(load_A32(ytmp, NS, ks * 32, lane), load_B32(w0h, NS, wv * 16, ks * 32, lane), acc);
        const int n = wv * 16 + bcol;
#pragma unroll
        for (int r = 0; r < 8; ++r)
          a1[(r + 8 * hh) * NH + n] = (_Float16)softplus_f(acc[r] + (float)b0h[n]);
      }
      __syncthreads();
      { // L2: h1 x W1^T -> h2 (softplus)
        v8f acc = {};
#pragma unroll
        for (int ks = 0; ks < 4; ++ks)
          acc = WMMA_F16(load_A32(a1, NH, ks * 32, lane), load_B32(w1h, NH, wv * 16, ks * 32, lane), acc);
        const int n = wv * 16 + bcol;
#pragma unroll
        for (int r = 0; r < 8; ++r)
          a2[(r + 8 * hh) * NH + n] = (_Float16)softplus_f(acc[r] + (float)b1h[n]);
      }
      __syncthreads();
      { // L3 + einsum: 8 N-tiles per wave; tile n-cols == c-dim for one s
        float dxv[8];
#pragma unroll
        for (int r = 0; r < 8; ++r) dxv[r] = dxs[(r + 8 * hh) * NC + bcol];
        v16h afr[4];
#pragma unroll
        for (int ks = 0; ks < 4; ++ks) afr[ks] = load_A32(a2, NH, ks * 32, lane);
#pragma unroll
        for (int i = 0; i < 8; ++i) {
          const int s = wv * 8 + i, n0 = s * NC;
          v8f acc = {};
#pragma unroll
          for (int ks = 0; ks < 4; ++ks)
            acc = WMMA_F16(afr[ks], load_B32(w2h, NH, n0, ks * 32, lane), acc);
          const float bb = (float)b2h[n0 + bcol];
#pragma unroll
          for (int r = 0; r < 8; ++r) {
            float v = fast_tanh(acc[r] + bb) * dxv[r];   // vf[b][s][c] * dX[b][c]
            v += __shfl_xor(v, 1);                        // reduce over c (16 lanes)
            v += __shfl_xor(v, 2);
            v += __shfl_xor(v, 4);
            v += __shfl_xor(v, 8);
            if (bcol == 0) fstg[(r + 8 * hh) * NS + s] = v;  // h1 region is free now
          }
        }
      }
      __syncthreads();
#pragma unroll
      for (int q = 0; q < 4; ++q) kk[st][q] = fstg[bcol * NS + wv * 8 + hh * 4 + q];
    }
    // y update (5th-order solution weights)
#pragma unroll
    for (int q = 0; q < 4; ++q)
      y[q] += hstep * (BWc[0] * kk[0][q] + BWc[1] * kk[1][q] + BWc[2] * kk[2][q] +
                       BWc[3] * kk[3][q] + BWc[4] * kk[4][q] + BWc[5] * kk[5][q]);
    readout(t + 1);
  }
}

extern "C" void kernel_launch(void* const* d_in, const int* in_sizes, int n_in,
                              void* d_out, int out_size, void* d_ws, size_t ws_size,
                              hipStream_t stream) {
  (void)in_sizes; (void)n_in; (void)out_size; (void)d_ws; (void)ws_size;
  const float* xs   = (const float*)d_in[0];
  const float* icw0 = (const float*)d_in[1];
  const float* icb0 = (const float*)d_in[2];
  const float* icw1 = (const float*)d_in[3];
  const float* icb1 = (const float*)d_in[4];
  const float* icw2 = (const float*)d_in[5];
  const float* icb2 = (const float*)d_in[6];
  const float* vfw0 = (const float*)d_in[7];
  const float* vfb0 = (const float*)d_in[8];
  const float* vfw1 = (const float*)d_in[9];
  const float* vfb1 = (const float*)d_in[10];
  const float* vfw2 = (const float*)d_in[11];
  const float* vfb2 = (const float*)d_in[12];
  const float* roW  = (const float*)d_in[13];
  const float* roB  = (const float*)d_in[14];
  float* out = (float*)d_out;

  (void)hipFuncSetAttribute((const void*)neural_cde_kernel,
                            hipFuncAttributeMaxDynamicSharedMemorySize, SMEM_BYTES);
  neural_cde_kernel<<<dim3(NB / 16), dim3(256), SMEM_BYTES, stream>>>(
      xs, icw0, icb0, icw1, icb1, icw2, icb2,
      vfw0, vfb0, vfw1, vfb1, vfw2, vfb2, roW, roB, out);
}
